// BiLSTMEncoder_16904991277118
// MI455X (gfx1250) — matrix operand order repaired and checked by reading, not compiled, not run on hardware
//
#include <hip/hip_runtime.h>

// ---------------------------------------------------------------------------
// BiLSTM encoder for MI455X (gfx1250, wave32).
// B=64, T=512, V=50000, E=256, H=512.
//
// Persistent recurrent kernel, one pass = 32 blocks (16/direction):
//   - block bq of dir d owns 2 h-column tiles x all 4 batch tiles
//   - weights (128 cols x K=768, Wih||Whh, bf16) staged in LDS ONCE (~194KB)
//   - z = [x_t ; h] panel (64 x 768 bf16) staged in LDS per step (~97KB),
//     via GLOBAL_LOAD_ASYNC_TO_LDS_B128 (ASYNCcnt) on this toolchain
//   - all WMMA operands from LDS -> zero weight traffic on the serial path
//   - per-direction device barrier (atomic counter), c/h carries in VGPRs
// ---------------------------------------------------------------------------

#define Bv 64
#define Tv 512
#define Ev 256
#define Hv 512
#define Kz 768            // E + H
#define PITCH 776         // Kz + 8 pad (LDS bank stagger)
#define UNK 1
#define BLK_PER_DIR 16
#define SMEM_BYTES ((128 * PITCH + 64 * PITCH) * 2)   // 297,984 B < 320 KB

typedef __attribute__((ext_vector_type(16))) __bf16 v16bf;
typedef __attribute__((ext_vector_type(8)))  __bf16 v8bf;
typedef __attribute__((ext_vector_type(8)))  float  v8f;
typedef __attribute__((ext_vector_type(4)))  int    v4i;

#define ASG __attribute__((address_space(1)))
#define ASL __attribute__((address_space(3)))

#if defined(__has_builtin)
#if __has_builtin(__builtin_amdgcn_global_load_async_to_lds_b128) && \
    __has_builtin(__builtin_amdgcn_s_wait_asynccnt)
#define HAVE_ASYNC_LDS 1
#endif
#endif

static __device__ __forceinline__ __bf16 f2bf(float f) { return (__bf16)f; }
static __device__ __forceinline__ float sigf(float x) { return 1.0f / (1.0f + __expf(-x)); }
static __device__ __forceinline__ v16bf mk16(v8bf lo, v8bf hi) {
    v16bf r;
    #pragma unroll
    for (int i = 0; i < 8; ++i) { r[i] = lo[i]; r[i + 8] = hi[i]; }
    return r;
}

// 16-byte global->LDS copy: async DMA on CDNA5 toolchains, sync fallback.
static __device__ __forceinline__ void g2l16(const __bf16* g, __bf16* l) {
#ifdef HAVE_ASYNC_LDS
    __builtin_amdgcn_global_load_async_to_lds_b128(
        (ASG v4i*)(void*)(const_cast<__bf16*>(g)),
        (ASL v4i*)(void*)(l), 0, 0);
#else
    *(v8bf*)l = *(const v8bf*)g;
#endif
}
static __device__ __forceinline__ void g2l_wait() {
#ifdef HAVE_ASYNC_LDS
    __builtin_amdgcn_s_wait_asynccnt(0);
#endif
}

// ---------------------------------------------------------------------------
__global__ void cvt_bf16_kernel(const float* __restrict__ src,
                                __bf16* __restrict__ dst, int n) {
    int i = blockIdx.x * blockDim.x + threadIdx.x;
    if (i < n) dst[i] = f2bf(src[i]);
}

// Embedding gather -> bf16.  grid = B*T blocks of E threads.
__global__ void embed_kernel(const int* __restrict__ ids,
                             const float* __restrict__ table,
                             __bf16* __restrict__ xb) {
    int bt = blockIdx.x;
    int id = ids[bt];
    xb[(long)bt * Ev + threadIdx.x] = f2bf(table[(long)id * Ev + threadIdx.x]);
}

// Zero h ping-pong buffer + the two per-direction barrier counters of a pass.
__global__ void zero_kernel(__bf16* __restrict__ hbuf, int n,
                            unsigned* __restrict__ ctrs) {
    int i = blockIdx.x * blockDim.x + threadIdx.x;
    if (i < n) hbuf[i] = (__bf16)0.0f;
    if (i < 2) ctrs[i] = 0u;
}

// UNK replacement: x[b,t,:] = Wp @ [h1[b,t-1,:H] ; h1[b,t+1,H:]] + bp
__global__ void unk_kernel(const int* __restrict__ ids,
                           const int* __restrict__ lens,
                           const float* __restrict__ h,   // pass-1 h (B,T,2H)
                           const float* __restrict__ Wp,  // (E, 2H)
                           const float* __restrict__ bp,  // (E)
                           __bf16* __restrict__ xb) {
    int bt = blockIdx.x;
    int b = bt / Tv, t = bt % Tv;
    int len = lens[b];
    if (!(ids[bt] == UNK && t < len)) return;

    __shared__ float ctx[2 * Hv];
    for (int k = threadIdx.x; k < 2 * Hv; k += blockDim.x) {
        float v;
        if (k < Hv) v = (t > 0) ? h[((long)(b * Tv + t - 1)) * (2 * Hv) + k] : 0.0f;
        else        v = (t + 1 < len) ? h[((long)(b * Tv + t + 1)) * (2 * Hv) + k] : 0.0f;
        ctx[k] = v;
    }
    __syncthreads();

    int e = threadIdx.x;            // blockDim == E == 256
    float acc = bp[e];
    const float* w = Wp + (long)e * (2 * Hv);
    #pragma unroll 8
    for (int k = 0; k < 2 * Hv; ++k) acc = fmaf(w[k], ctx[k], acc);
    xb[(long)bt * Ev + e] = f2bf(acc);
}

// ---------------------------------------------------------------------------
// Persistent bidirectional LSTM pass.
//   grid = 32 blocks x 256 threads. dir = blockIdx.x>>4, bq = blockIdx.x&15.
//   wave w (0..7): nt = w>>2 (local n-tile), m0l = (w&3)*16 (batch tile).
//   LDS: wsh[128][776] weight cols (nt,g,c) staged once;
//        zsh[64][776]  z=[x_t;h] panel staged per step.
// ---------------------------------------------------------------------------
__global__ void __launch_bounds__(256)
lstm_pass_kernel(const __bf16* __restrict__ xb,
                 const __bf16* __restrict__ WihF, const __bf16* __restrict__ WhhF,
                 const __bf16* __restrict__ WihB, const __bf16* __restrict__ WhhB,
                 const float* __restrict__ bF, const float* __restrict__ bB,
                 const int* __restrict__ lens,
                 __bf16* __restrict__ hbuf,      // 2 bufs x 2 dirs x B x H
                 unsigned* __restrict__ ctrs,    // [2] per-direction counters
                 float* __restrict__ out) {      // h | h_n | c_n
    extern __shared__ char smem[];
    __bf16* wsh = (__bf16*)smem;                          // 128 x PITCH
    __bf16* zsh = (__bf16*)(smem + 128 * PITCH * 2);      //  64 x PITCH

    const int dir  = blockIdx.x >> 4;
    const int bq   = blockIdx.x & 15;
    const int wave = threadIdx.x >> 5;
    const int lane = threadIdx.x & 31;
    const int nt   = wave >> 2;              // local n-tile 0..1
    const int m0l  = (wave & 3) * 16;        // batch-tile base
    const int n0   = (bq * 2 + nt) * 16;     // h-column tile base
    const int col  = n0 + (lane & 15);
    const int hi   = lane >> 4;
    const int aofs = hi ? 8 : 0;             // A-fragment K split (ISA layout)
    const int bofs = hi ? 8 : 0;             // B-fragment K split

    const __bf16* Wih  = dir ? WihB : WihF;  // (4H, E) row-major, bf16
    const __bf16* Whh  = dir ? WhhB : WhhF;  // (4H, H) row-major, bf16
    const float*  bias = dir ? bB : bF;
    unsigned* ctr = ctrs + dir;

    // ---- stage this block's 128 weight columns into LDS, once ----
    // wsh row (nt2*4+g)*16+c holds [Wih[gcol,:] ; Whh[gcol,:]] (K=768)
    for (int idx = threadIdx.x; idx < 128 * (Kz / 8); idx += 256) {
        int row = idx / (Kz / 8);
        int kk  = (idx % (Kz / 8)) * 8;
        int nt2 = row >> 6, g = (row >> 4) & 3, c = row & 15;
        int gcol = g * Hv + (bq * 2 + nt2) * 16 + c;
        const __bf16* src = (kk < Ev) ? (Wih + (long)gcol * Ev + kk)
                                      : (Whh + (long)gcol * Hv + (kk - Ev));
        g2l16(src, wsh + row * PITCH + kk);
    }

    float bv[4];
    #pragma unroll
    for (int g = 0; g < 4; ++g) bv[g] = bias[g * Hv + col];

    int rows[8], len8[8];
    float cst[8], hst[8];
    #pragma unroll
    for (int i = 0; i < 8; ++i) {
        rows[i] = m0l + i + hi * 8;          // C/D layout: M = i (+8 hi lanes)
        len8[i] = lens[rows[i]];
        cst[i] = 0.0f; hst[i] = 0.0f;
    }

    const __bf16* abase = zsh + (m0l + (lane & 15)) * PITCH + aofs;
    const __bf16* bbase[4];
    #pragma unroll
    for (int g = 0; g < 4; ++g)
        bbase[g] = wsh + ((nt * 4 + g) * 16 + (lane & 15)) * PITCH + bofs;

    const int HB = 2 * Bv * Hv;              // elems per ping-pong buffer

    for (int ts = 0; ts < Tv; ++ts) {
        const int t = dir ? (Tv - 1 - ts) : ts;
        const __bf16* hread  = hbuf + (long)(ts & 1) * HB + (long)dir * Bv * Hv;
        __bf16*       hwrite = hbuf + (long)((ts + 1) & 1) * HB + (long)dir * Bv * Hv;

        // ---- stage z = [x_t ; h] for all 64 batch rows (async -> LDS) ----
        for (int c = threadIdx.x; c < Bv * (Kz / 8); c += 256) {
            int r = c / (Kz / 8), kk = (c % (Kz / 8)) * 8;
            const __bf16* src = (kk < Ev)
                ? (xb + ((long)r * Tv + t) * Ev + kk)
                : (hread + (long)r * Hv + (kk - Ev));
            g2l16(src, zsh + r * PITCH + kk);
        }
        // prefetch next timestep's x rows (global_prefetch_b8)
        if (ts + 1 < Tv && threadIdx.x < Bv) {
            int t2 = dir ? (Tv - 2 - ts) : (ts + 1);
            __builtin_prefetch(xb + ((long)threadIdx.x * Tv + t2) * Ev, 0, 1);
        }
        g2l_wait();
        __syncthreads();

        // ---- gates = z @ [Wih;Whh]^T + b, all operands from LDS ----
        v8f acc[4];
        #pragma unroll
        for (int g = 0; g < 4; ++g)
            #pragma unroll
            for (int i = 0; i < 8; ++i) acc[g][i] = bv[g];

        #pragma unroll 4
        for (int kc = 0; kc < Kz / 32; ++kc) {
            const int k0 = kc * 32;
            v16bf a = mk16(*(const v8bf*)(abase + k0),
                           *(const v8bf*)(abase + k0 + 16));
            #pragma unroll
            for (int g = 0; g < 4; ++g) {
                v16bf bf = mk16(*(const v8bf*)(bbase[g] + k0),
                                *(const v8bf*)(bbase[g] + k0 + 16));
                acc[g] = __builtin_amdgcn_wmma_f32_16x16x32_bf16(
                    false, a, false, bf, (short)0, acc[g], false, false);
            }
        }

        // ---- elementwise LSTM cell, masked carry, outputs ----
        #pragma unroll
        for (int i = 0; i < 8; ++i) {
            float ig = sigf(acc[0][i]);
            float fg = sigf(acc[1][i]);
            float gg = tanhf(acc[2][i]);
            float og = sigf(acc[3][i]);
            float cn = fg * cst[i] + ig * gg;
            float hn = og * tanhf(cn);
            bool  m  = (t < len8[i]);
            cst[i] = m ? cn : cst[i];
            hst[i] = m ? hn : hst[i];
            out[((long)rows[i] * Tv + t) * (2 * Hv) + dir * Hv + col] = m ? hn : 0.0f;
            hwrite[(long)rows[i] * Hv + col] = f2bf(hst[i]);
        }

        // ---- per-direction device barrier (skip after last step) ----
        if (ts + 1 < Tv) {
            __threadfence();
            __syncthreads();
            if (threadIdx.x == 0) {
                __hip_atomic_fetch_add(ctr, 1u, __ATOMIC_RELEASE,
                                       __HIP_MEMORY_SCOPE_AGENT);
                unsigned want = (unsigned)(ts + 1) * (unsigned)BLK_PER_DIR;
                while (__hip_atomic_load(ctr, __ATOMIC_ACQUIRE,
                                         __HIP_MEMORY_SCOPE_AGENT) < want)
                    __builtin_amdgcn_s_sleep(1);
            }
            __syncthreads();
            __threadfence();
        }
    }

    // ---- h_n, c_n from register carries ----
    const long hb = (long)Bv * Tv * 2 * Hv;
    #pragma unroll
    for (int i = 0; i < 8; ++i) {
        out[hb + (long)dir * Bv * Hv + (long)rows[i] * Hv + col] = hst[i];
        out[hb + 2L * Bv * Hv + (long)dir * Bv * Hv + (long)rows[i] * Hv + col] = cst[i];
    }
}

// ---------------------------------------------------------------------------
extern "C" void kernel_launch(void* const* d_in, const int* in_sizes, int n_in,
                              void* d_out, int out_size, void* d_ws, size_t ws_size,
                              hipStream_t stream) {
    const int*   enc  = (const int*)  d_in[0];
    const int*   lens = (const int*)  d_in[1];
    const float* emb  = (const float*)d_in[2];
    const float* WihF = (const float*)d_in[3];
    const float* WhhF = (const float*)d_in[4];
    const float* bF   = (const float*)d_in[5];
    const float* WihB = (const float*)d_in[6];
    const float* WhhB = (const float*)d_in[7];
    const float* bB   = (const float*)d_in[8];
    const float* Wp   = (const float*)d_in[9];
    const float* bp   = (const float*)d_in[10];
    float* out = (float*)d_out;

    char* ws = (char*)d_ws;
    __bf16* xb    = (__bf16*)(ws);                    // B*T*E bf16     = 16 MiB
    __bf16* wihf  = (__bf16*)(ws + 16777216);
    __bf16* whhf  = (__bf16*)(ws + 17825792);
    __bf16* wihb  = (__bf16*)(ws + 19922944);
    __bf16* whhb  = (__bf16*)(ws + 20971520);
    __bf16* hbuf  = (__bf16*)(ws + 23068672);         // 2*2*B*H bf16   = 256 KiB
    unsigned* ctrP0 = (unsigned*)(ws + 23330816);     // pass-1 dir counters [2]
    unsigned* ctrP1 = (unsigned*)(ws + 23331072);     // pass-2 dir counters [2]

    const int nWih = 4 * Hv * Ev;
    const int nWhh = 4 * Hv * Hv;
    const int nHbuf = 2 * 2 * Bv * Hv;

    // allow ~291 KB dynamic LDS on the CDNA5 WGP (320 KB available)
    (void)hipFuncSetAttribute((const void*)lstm_pass_kernel,
                              hipFuncAttributeMaxDynamicSharedMemorySize,
                              SMEM_BYTES);

    cvt_bf16_kernel<<<(nWih + 255) / 256, 256, 0, stream>>>(WihF, wihf, nWih);
    cvt_bf16_kernel<<<(nWhh + 255) / 256, 256, 0, stream>>>(WhhF, whhf, nWhh);
    cvt_bf16_kernel<<<(nWih + 255) / 256, 256, 0, stream>>>(WihB, wihb, nWih);
    cvt_bf16_kernel<<<(nWhh + 255) / 256, 256, 0, stream>>>(WhhB, whhb, nWhh);

    embed_kernel<<<Bv * Tv, Ev, 0, stream>>>(enc, emb, xb);

    // ---- pass 1 ----
    zero_kernel<<<(nHbuf + 255) / 256, 256, 0, stream>>>(hbuf, nHbuf, ctrP0);
    lstm_pass_kernel<<<32, 256, SMEM_BYTES, stream>>>(
        xb, wihf, whhf, wihb, whhb, bF, bB, lens, hbuf, ctrP0, out);

    // ---- UNK embedding replacement using pass-1 h (in d_out) ----
    unk_kernel<<<Bv * Tv, 256, 0, stream>>>(enc, lens, out, Wp, bp, xb);

    // ---- pass 2 (overwrites h, writes final h_n / c_n) ----
    zero_kernel<<<(nHbuf + 255) / 256, 256, 0, stream>>>(hbuf, nHbuf, ctrP1);
    lstm_pass_kernel<<<32, 256, SMEM_BYTES, stream>>>(
        xb, wihf, whhf, wihb, whhb, bF, bB, lens, hbuf, ctrP1, out);
}
